// Attention_61048665145392
// MI455X (gfx1250) — matrix operand order repaired and checked by reading, not compiled, not run on hardware
//
#include <hip/hip_runtime.h>

#define B_BATCH 4
#define S_LEN   4096
#define W_SIZE  512
#define E_DIM   64

typedef __attribute__((ext_vector_type(16))) _Float16 v16h;
typedef __attribute__((ext_vector_type(8)))  _Float16 v8h;
typedef __attribute__((ext_vector_type(8)))  float    v8f;

#if __has_builtin(__builtin_amdgcn_exp2f)
#define EXP2F(x) __builtin_amdgcn_exp2f(x)
#else
#define EXP2F(x) exp2f(x)
#endif

union HU { unsigned u; _Float16 h[2]; };

__device__ __forceinline__ unsigned pack2(_Float16 a, _Float16 b) {
  HU t; t.h[0] = a; t.h[1] = b; return t.u;
}

__device__ __forceinline__ v8f wmma_f16(v16h a, v16h b, v8f c) {
  // (neg_a, A, neg_b, B, c_mod, C, reuse_a, reuse_b)
  return __builtin_amdgcn_wmma_f32_16x16x32_f16(false, a, false, b, (short)0, c,
                                                false, false);
}

// Async 16B/lane copy global -> LDS (CDNA5, tracked by ASYNCcnt).
// lds_off = workgroup-relative LDS byte offset (low 32 bits of flat shared ptr).
__device__ __forceinline__ void async_load_b128(unsigned lds_off, const void* gaddr) {
  asm volatile("global_load_async_to_lds_b128 %0, %1, off"
               :: "v"(lds_off), "v"(gaddr) : "memory");
}
__device__ __forceinline__ void wait_asynccnt0() {
  asm volatile("s_wait_asynccnt 0x0" ::: "memory");
}

// ---- fp32-source fragments (projection kernel) ----------------------------
// A (16x32 f16, lane = M): elem j -> K = c0 + 16h + j
__device__ __forceinline__ v16h load_frag_a_f32(const float* __restrict__ rp, int c0, int h) {
  const float* p = rp + c0 + 16 * h;
  v16h f;
#pragma unroll
  for (int j = 0; j < 8; ++j) f[j] = (_Float16)p[j];
#pragma unroll
  for (int j = 0; j < 8; ++j) f[8 + j] = (_Float16)p[8 + j];
  return f;
}
// B (32x16 f16, lane = N): elem j -> K = c0 + 16h + j
__device__ __forceinline__ v16h load_frag_b_f32(const float* __restrict__ rp, int c0, int h) {
  const float* p = rp + c0 + 16 * h;
  v16h f;
#pragma unroll
  for (int j = 0; j < 16; ++j) f[j] = (_Float16)p[j];
  return f;
}

// ---- f16-source A fragment: two aligned 16B loads (global or LDS) ---------
__device__ __forceinline__ v16h load_frag_a_h(const _Float16* rp, int c0, int h) {
  const _Float16* p = rp + c0 + 16 * h;
  const v8h lo = *(const v8h*)p;
  const v8h hi = *(const v8h*)(p + 8);
  v16h f;
#pragma unroll
  for (int j = 0; j < 8; ++j) { f[j] = lo[j]; f[8 + j] = hi[j]; }
  return f;
}

// ---------------- Kernel 1: QKV projection + precision/layout staging ------
// grid = (M/16, 3), block = 128 (4 waves). Wave w computes a 16x16 tile of
// (x @ W^T + b). Q,K stored f16 row-major [B*S][E]; V stored f16 transposed
// [B][E][S] so the attention kernel's V^T A-fragments are contiguous.
__global__ void __launch_bounds__(128) qkv_proj_kernel(
    const float* __restrict__ x,
    const float* __restrict__ Wq, const float* __restrict__ bq,
    const float* __restrict__ Wk, const float* __restrict__ bk,
    const float* __restrict__ Wv, const float* __restrict__ bv,
    _Float16* __restrict__ Qh, _Float16* __restrict__ Kh,
    _Float16* __restrict__ Vt)
{
  const int lane = threadIdx.x & 31;
  const int wave = threadIdx.x >> 5;
  const int h  = lane >> 4;
  const int nl = lane & 15;
  const int m0 = blockIdx.x * 16;
  const int n0 = wave * 16;

  const float* Wm; const float* bm;
  if (blockIdx.y == 0)      { Wm = Wq; bm = bq; }
  else if (blockIdx.y == 1) { Wm = Wk; bm = bk; }
  else                      { Wm = Wv; bm = bv; }

  const float* xrow = x  + (size_t)(m0 + nl) * W_SIZE;   // A: M = token row
  const float* wrow = Wm + (size_t)(n0 + nl) * W_SIZE;   // B: N = out-dim

  v8f acc = {};
#pragma unroll 4
  for (int c0 = 0; c0 < W_SIZE; c0 += 32) {
    v16h a = load_frag_a_f32(xrow, c0, h);
    v16h b = load_frag_b_f32(wrow, c0, h);
    acc = wmma_f16(a, b, acc);
  }

  const float bias = bm[n0 + nl];
  if (blockIdx.y != 2) {                   // Q or K: row-major [m][e] f16
    _Float16* om = (blockIdx.y == 0) ? Qh : Kh;
#pragma unroll
    for (int r = 0; r < 8; ++r)
      om[(size_t)(m0 + 8 * h + r) * E_DIM + n0 + nl] = (_Float16)(acc[r] + bias);
  } else {                                 // V: transposed [b][e][s] f16
    const int bb = m0 / S_LEN;
    const int s0 = (m0 % S_LEN) + 8 * h;   // regs r -> 8 consecutive s
    v8h val;
#pragma unroll
    for (int r = 0; r < 8; ++r) val[r] = (_Float16)(acc[r] + bias);
    *(v8h*)(Vt + ((size_t)bb * E_DIM + n0 + nl) * S_LEN + s0) = val;
  }
}

// ---------------- Kernel 2: flash attention ------------------------------
// grid = (B * S/16), block = 128 (4 waves). Block = one (batch, 16-query)
// tile; wave w owns output dims [16w,16w+16).
//   scoreT = K_tile * Q^T  (softmax reduce = 8 regs + one lane^16 shuffle)
//   Z^T    = V^T * P^T     (accumulator rescale uniform per lane)
// The 4KB K tile is shared by all 4 waves, so it is staged into LDS once per
// block with async global->LDS b128 copies (ASYNCcnt), double-buffered so the
// copy of tile i+1 overlaps WMMA compute on tile i.
__global__ void __launch_bounds__(128) flash_attn_kernel(
    const _Float16* __restrict__ Q, const _Float16* __restrict__ K,
    const _Float16* __restrict__ Vt, float* __restrict__ out)
{
  __shared__ alignas(16) _Float16 Kbuf[2][32 * E_DIM];  // 2 x 4KB K tiles
  __shared__ unsigned Plds[4][16 * 16];    // per-wave 16(q) x 32(k) f16 pairs

  const int tid  = threadIdx.x;
  const int lane = tid & 31;
  const int wave = tid >> 5;
  const int h  = lane >> 4;
  const int nl = lane & 15;
  const int b  = blockIdx.x >> 8;          // S/16 = 256 query tiles per batch
  const int q0 = (blockIdx.x & 255) * 16;

  const _Float16* Qb = Q  + (size_t)b * S_LEN * E_DIM;
  const _Float16* Kb = K  + (size_t)b * S_LEN * E_DIM;
  const _Float16* Vb = Vt + (size_t)b * E_DIM * S_LEN;

  // Q as B-operand: B[e][q] = Q[q][e]; elem j -> e = c0 + 16h + j (32B aligned)
  const _Float16* qrow = Qb + (size_t)(q0 + nl) * E_DIM;
  const v16h bq0 = *(const v16h*)(qrow + 16 * h);
  const v16h bq1 = *(const v16h*)(qrow + 32 + 16 * h);

  unsigned* pw = &Plds[wave][0];
  const _Float16* vrow = Vb + (size_t)(wave * 16 + nl) * S_LEN; // V^T row e

  // Async K staging: 128 threads x 16B x 2 = 4KB tile (2 instrs per wave).
  const int srow = tid >> 3;               // 0..15
  const int scol = (tid & 7) * 8;          // f16 chunk start within a row
  auto stage = [&](int p, int kb) {
    async_load_b128((unsigned)(size_t)&Kbuf[p][srow * E_DIM + scol],
                    Kb + (size_t)(kb + srow) * E_DIM + scol);
    async_load_b128((unsigned)(size_t)&Kbuf[p][(16 + srow) * E_DIM + scol],
                    Kb + (size_t)(kb + 16 + srow) * E_DIM + scol);
  };
  stage(0, 0);

  v8f acc = {};
  float mrow = -1.0e30f;                   // running max (log2 domain), query nl
  float lrow = 0.0f;                       // running sum
  const float scale2 = 0.125f * 1.44269504088896340736f; // (1/sqrt(64))*log2(e)

  for (int kb = 0, it = 0; kb < S_LEN; kb += 32, ++it) {
    const int p = it & 1;
    wait_asynccnt0();                      // my buf[p] pieces arrived
    __syncthreads();                       // all waves' pieces arrived; prev
                                           // reads of buf[p^1] are done
    if (kb + 32 < S_LEN) stage(p ^ 1, kb + 32);  // overlap with compute

    // Two 16x16 scoreT tiles from LDS (keys kb..kb+15, kb+16..kb+31).
    const _Float16* kr0 = &Kbuf[p][nl * E_DIM];
    const _Float16* kr1 = &Kbuf[p][(16 + nl) * E_DIM];
    v8f s0 = {}, s1 = {};
    s0 = wmma_f16(load_frag_a_h(kr0, 0, h),  bq0, s0);
    s0 = wmma_f16(load_frag_a_h(kr0, 32, h), bq1, s0);
    s1 = wmma_f16(load_frag_a_h(kr1, 0, h),  bq0, s1);
    s1 = wmma_f16(load_frag_a_h(kr1, 32, h), bq1, s1);

    if (kb + 32 < S_LEN)                   // uniform branch
      __builtin_prefetch(vrow + kb + 32, 0, 1);

    // Online softmax (log2 domain). s0[r]=scoreT[8h+r][nl], s1[r]=scoreT[16+8h+r][nl].
    float pm = -1.0e30f;
#pragma unroll
    for (int r = 0; r < 8; ++r) {
      s0[r] *= scale2; s1[r] *= scale2;
      pm = fmaxf(pm, fmaxf(s0[r], s1[r]));
    }
    pm = fmaxf(pm, __shfl_xor(pm, 16, 32));
    const float mnew = fmaxf(mrow, pm);
    const float corr = EXP2F(mrow - mnew);

    float p0[8], p1[8];
    float ls = 0.0f;
#pragma unroll
    for (int r = 0; r < 8; ++r) {
      p0[r] = EXP2F(s0[r] - mnew);
      p1[r] = EXP2F(s1[r] - mnew);
      ls += p0[r] + p1[r];
    }
    ls += __shfl_xor(ls, 16, 32);
    lrow = lrow * corr + ls;
    mrow = mnew;
#pragma unroll
    for (int r = 0; r < 8; ++r) acc[r] *= corr;   // per-lane uniform rescale

    // Stage P^T through LDS: row q = nl, 16 uints (32 f16) along k.
#pragma unroll
    for (int j = 0; j < 4; ++j) {
      pw[nl * 16 + 4 * h + j]     = pack2((_Float16)p0[2 * j], (_Float16)p0[2 * j + 1]);
      pw[nl * 16 + 8 + 4 * h + j] = pack2((_Float16)p1[2 * j], (_Float16)p1[2 * j + 1]);
    }

    // Reload as B-operand of P^T: elem j -> k = 16h + j (uint index 8h + j/2).
    v16h bp;
#pragma unroll
    for (int j = 0; j < 8; ++j) {
      HU t; t.u = pw[nl * 16 + 8 * h + j];
      bp[2 * j]     = t.h[0];
      bp[2 * j + 1] = t.h[1];
    }

    // A-operand of V^T: row e, contraction k — contiguous f16, 2x b128.
    v16h av = load_frag_a_h(vrow + kb, 0, h);

    acc = wmma_f16(av, bp, acc);           // Z^T accumulate, K = 32 keys
  }

  const float inv = 1.0f / lrow;
  float* orow = out + ((size_t)b * S_LEN + q0 + nl) * E_DIM + wave * 16;
#pragma unroll
  for (int r = 0; r < 8; ++r) orow[8 * h + r] = acc[r] * inv;
}

// ---------------- Launch ---------------------------------------------------
extern "C" void kernel_launch(void* const* d_in, const int* in_sizes, int n_in,
                              void* d_out, int out_size, void* d_ws, size_t ws_size,
                              hipStream_t stream) {
  (void)in_sizes; (void)n_in; (void)out_size; (void)ws_size;
  const float* x  = (const float*)d_in[0];
  const float* Wq = (const float*)d_in[1];
  const float* bq = (const float*)d_in[2];
  const float* Wk = (const float*)d_in[3];
  const float* bk = (const float*)d_in[4];
  const float* Wv = (const float*)d_in[5];
  const float* bv = (const float*)d_in[6];
  float* out = (float*)d_out;

  const size_t perMat = (size_t)B_BATCH * S_LEN * E_DIM;  // 1 Mi elements
  _Float16* Qh = (_Float16*)d_ws;
  _Float16* Kh = Qh + perMat;
  _Float16* Vt = Kh + perMat;   // [B][E][S] layout

  dim3 gProj((B_BATCH * S_LEN) / 16, 3);
  qkv_proj_kernel<<<gProj, 128, 0, stream>>>(x, Wq, bq, Wk, bk, Wv, bv, Qh, Kh, Vt);

  dim3 gAttn((B_BATCH * S_LEN) / 16);
  flash_attn_kernel<<<gAttn, 128, 0, stream>>>(Qh, Kh, Vt, out);
}
